// GraphAttentionLayer_12266426598001
// MI455X (gfx1250) — matrix operand order repaired and checked by reading, not compiled, not run on hardware
//
#include <hip/hip_runtime.h>
#include <hip/hip_bf16.h>
#include <stdint.h>

#define NNODES 16384
#define MNEI   32
#define F      256
#define ALPHA  0.2f
#define LDSB   264   // bf16 tile row stride (264*2B = 132 dwords, %64 = 4 -> rows hit distinct banks)

typedef __attribute__((ext_vector_type(16))) __bf16 v16bf;
typedef __attribute__((ext_vector_type(8)))  __bf16 v8bf;
typedef __attribute__((ext_vector_type(8)))  float  v8f;
typedef __attribute__((ext_vector_type(4)))  unsigned int uint32x4;
typedef __attribute__((ext_vector_type(8)))  int    int32x8;
typedef __attribute__((ext_vector_type(4)))  int    int32x4;

// -------- prep: W_top transposed -> bf16 (Wt[o][k] = W[k][o]) ----------
__global__ void prep_wt_kernel(const float* __restrict__ W, __bf16* __restrict__ Wt) {
    int idx = blockIdx.x * 256 + threadIdx.x;   // 0 .. 65535
    int o = idx >> 8;
    int k = idx & 255;
    Wt[idx] = (__bf16)W[k * F + o];
}

// -------- prep: q[n][o] = sum_k output[n][k] * W_bot[k][o] -------------
__global__ __launch_bounds__(256) void compute_q_kernel(const float* __restrict__ output,
                                                        const float* __restrict__ W,
                                                        float* __restrict__ q) {
    __shared__ float so[32][F];
    const int nb  = blockIdx.x * 32;
    const int tid = threadIdx.x;

    for (int i = tid; i < 32 * F / 4; i += 256) {
        int row = (i * 4) / F, col = (i * 4) % F;
        float4 v = ((const float4*)(output + (size_t)nb * F))[i];
        *(float4*)&so[row][col] = v;
    }
    __syncthreads();

    float acc[32];
#pragma unroll
    for (int i = 0; i < 32; ++i) acc[i] = 0.f;

    const float* wb = W + (size_t)F * F;  // W_bot rows start at k=256
    const int o = tid;
    for (int k = 0; k < F; ++k) {
        float w = wb[(size_t)k * F + o];
#pragma unroll
        for (int i = 0; i < 32; ++i) acc[i] = fmaf(so[i][k], w, acc[i]);
    }
#pragma unroll
    for (int i = 0; i < 32; ++i) q[(size_t)(nb + i) * F + o] = acc[i];
}

// TDM: 1-D copy of `ndw` dwords from global to LDS (D# per CDNA5 ISA)
__device__ inline void tdm_load_1d_f32(uint32_t lds_off, const void* gptr, uint32_t ndw) {
    uint64_t ga = (uint64_t)(uintptr_t)gptr;
    uint32x4 g0;
    g0[0] = 1u;                                            // count=1, user-mode load
    g0[1] = lds_off;                                       // lds_addr (bytes)
    g0[2] = (uint32_t)(ga & 0xFFFFFFFFu);                  // global_addr[31:0]
    g0[3] = (uint32_t)((ga >> 32) & 0x01FFFFFFu)           // global_addr[56:32]
          | 0x80000000u;                                   // type=2 ("image") in bits[127:126]
    int32x8 g1;
    g1[0] = (int)(2u << 16);                               // data_size=2 -> 4-byte elements
    g1[1] = (int)((ndw & 0xFFFFu) << 16);                  // tensor_dim0[15:0]
    g1[2] = (int)((ndw >> 16) & 0xFFFFu) | (1 << 16);      // tensor_dim0[31:16] | tensor_dim1=1
    g1[3] = (int)((ndw & 0xFFFFu) << 16);                  // tile_dim0 (16-bit)
    g1[4] = 1;                                             // tile_dim1=1, tile_dim2=0
    g1[5] = (int)ndw;                                      // tensor_dim0_stride lo32
    g1[6] = 0;
    g1[7] = 0;
    int32x4 gz = {0, 0, 0, 0};                             // groups 2/3: dims 3/4 unused
#if __has_include(<hip/amd_detail/amd_gfx1250_TDM.h>)
    // amdgpu-toolchain (clang-23 / therock-10.0): 6-arg form
    int32x8 gz8 = {0, 0, 0, 0, 0, 0, 0, 0};
    __builtin_amdgcn_tensor_load_to_lds(g0, g1, gz, gz, gz8, 0);
#else
    // ROCm 7.2 (clang-22): 5-arg form
    __builtin_amdgcn_tensor_load_to_lds(g0, g1, gz, gz, 0);
#endif
}

__device__ inline v16bf frag_pair(const __bf16* base) {    // {base[0..7], base[16..23]}
    v8bf lo = *(const v8bf*)base;
    v8bf hi = *(const v8bf*)(base + 16);
    return __builtin_shufflevector(lo, hi, 0, 1, 2, 3, 4, 5, 6, 7,
                                           8, 9, 10, 11, 12, 13, 14, 15);
}

// -------- main fused GAT kernel: one block per node --------------------
__global__ __launch_bounds__(256, 2)
void gat_main_kernel(const float* __restrict__ transformed,   // [N][32][256] f32
                     const float* __restrict__ q,             // [N][256]     f32
                     const __bf16* __restrict__ Wt,           // [256][256]   bf16 (W_top^T)
                     const float* __restrict__ avec,          // [256]        f32
                     float* __restrict__ out_h,               // [N][256]
                     float* __restrict__ out_attn)            // [N][32]
{
    __shared__ float  Af[MNEI * F];       // f32 tile (TDM destination), unpadded
    __shared__ __bf16 Ab[MNEI * LDSB];    // bf16 tile in WMMA-friendly padded layout
    __shared__ float  e_sh[MNEI];
    __shared__ float  attn_sh[MNEI];

    const int n    = blockIdx.x;
    const int tid  = threadIdx.x;
    const int wave = tid >> 5;            // 0..7, owns output columns [32w, 32w+32)
    const int lane = tid & 31;
    const int l16  = lane & 15;
    const int hi   = lane >> 4;           // half-wave select

    if (tid < MNEI) e_sh[tid] = 0.f;

    // --- async-tensor stage of transformed[n] (32x256 f32 = 32 KB) into LDS ---
    if (wave == 0) {
        uint32_t lds_off = (uint32_t)(uintptr_t)&Af[0];    // LDS aperture: low 32 bits = offset
        tdm_load_1d_f32(lds_off, transformed + (size_t)n * MNEI * F, MNEI * F);
        __builtin_amdgcn_s_wait_tensorcnt(0);
    }
    __syncthreads();

    // --- one-shot f32 -> bf16 conversion into the padded matmul tile ---
    for (int c = tid; c < MNEI * F / 8; c += 256) {        // 8-element chunks
        int row = c >> 5;
        int col = (c & 31) * 8;
        const float* p = &Af[row * F + col];
        v8bf v;
#pragma unroll
        for (int i = 0; i < 8; ++i) v[i] = (__bf16)p[i];
        *(v8bf*)&Ab[row * LDSB + col] = v;
    }
    __syncthreads();

    // --- GEMM: (32x256) @ (256x256) restricted to this wave's 32 columns ---
    v8f acc[2][2];
#pragma unroll
    for (int mt = 0; mt < 2; ++mt)
#pragma unroll
        for (int nt = 0; nt < 2; ++nt)
#pragma unroll
            for (int r = 0; r < 8; ++r) acc[mt][nt][r] = 0.f;

    for (int kb = 0; kb < 8; ++kb) {                       // K blocks of 32
        v16bf afrag[2];
#pragma unroll
        for (int mt = 0; mt < 2; ++mt)                     // A: K{0..7,16..23} / {8..15,24..31}
            afrag[mt] = frag_pair(&Ab[(mt * 16 + l16) * LDSB + kb * 32 + hi * 8]);
#pragma unroll
        for (int nt = 0; nt < 2; ++nt) {                   // B: K{0..7,16..23} / {8..15,24..31}
            int col = wave * 32 + nt * 16 + l16;
            v16bf bfrag = frag_pair(Wt + (size_t)col * F + kb * 32 + hi * 8);
#pragma unroll
            for (int mt = 0; mt < 2; ++mt) {
                acc[mt][nt] = __builtin_amdgcn_wmma_f32_16x16x32_bf16(
                    false, afrag[mt], false, bfrag, (short)0, acc[mt][nt], false, false);
            }
        }
    }

    // --- fused epilogue: +q, leaky_relu, dot with a, reduce over columns ---
#pragma unroll
    for (int mt = 0; mt < 2; ++mt) {
        float p[8];
#pragma unroll
        for (int r = 0; r < 8; ++r) p[r] = 0.f;
#pragma unroll
        for (int nt = 0; nt < 2; ++nt) {
            int col = wave * 32 + nt * 16 + l16;
            float qv = q[(size_t)n * F + col];
            float av = avec[col];
#pragma unroll
            for (int r = 0; r < 8; ++r) {
                float v = acc[mt][nt][r] + qv;
                v = (v > 0.f) ? v : ALPHA * v;             // leaky_relu
                p[r] = fmaf(v, av, p[r]);
            }
        }
#pragma unroll
        for (int mask = 1; mask <= 8; mask <<= 1)          // sum 16 columns in half-wave
#pragma unroll
            for (int r = 0; r < 8; ++r) p[r] += __shfl_xor(p[r], mask, 32);
        if (l16 == 0) {
#pragma unroll
            for (int r = 0; r < 8; ++r)
                atomicAdd(&e_sh[mt * 16 + hi * 8 + r], p[r]);
        }
    }
    __syncthreads();

    // --- softmax over the 32 neighbors (wave 0) ---
    if (wave == 0) {
        float ev = e_sh[lane];
        float mx = ev;
#pragma unroll
        for (int mask = 16; mask >= 1; mask >>= 1) mx = fmaxf(mx, __shfl_xor(mx, mask, 32));
        float ex = __expf(ev - mx);
        float sm = ex;
#pragma unroll
        for (int mask = 16; mask >= 1; mask >>= 1) sm += __shfl_xor(sm, mask, 32);
        float at = ex / sm;
        attn_sh[lane] = at;
        out_attn[(size_t)n * MNEI + lane] = at;
    }
    __syncthreads();

    // --- h' = sum_m attn[m] * transformed[n,m,:] (f32 from LDS), then elu ---
    float h = 0.f;
#pragma unroll 8
    for (int m = 0; m < MNEI; ++m) h = fmaf(attn_sh[m], Af[m * F + tid], h);
    out_h[(size_t)n * F + tid] = (h > 0.f) ? h : (__expf(h) - 1.f);
}

extern "C" void kernel_launch(void* const* d_in, const int* in_sizes, int n_in,
                              void* d_out, int out_size, void* d_ws, size_t ws_size,
                              hipStream_t stream) {
    const float* transformed = (const float*)d_in[0];   // [N,32,256]
    const float* output      = (const float*)d_in[1];   // [N,256]
    const float* W           = (const float*)d_in[2];   // [512,256]
    const float* avec        = (const float*)d_in[3];   // [256]

    float* out_h    = (float*)d_out;                          // [N,256]
    float* out_attn = out_h + (size_t)NNODES * F;             // [N,32]

    float*  q  = (float*)d_ws;                                               // 16 MB
    __bf16* Wt = (__bf16*)((char*)d_ws + (size_t)NNODES * F * sizeof(float)); // 128 KB

    prep_wt_kernel<<<F * F / 256, 256, 0, stream>>>(W, Wt);
    compute_q_kernel<<<NNODES / 32, 256, 0, stream>>>(output, W, q);
    gat_main_kernel<<<NNODES, 256, 0, stream>>>(transformed, q, Wt, avec, out_h, out_attn);
}
